// ReachabilityFeaturesGNN_49185965474415
// MI455X (gfx1250) — matrix operand order repaired and checked by reading, not compiled, not run on hardware
//
#include <hip/hip_runtime.h>
#include <hip/hip_bf16.h>

// ---------------------------------------------------------------------------
// ReachabilityFeaturesGNN for MI455X (gfx1250, wave32).
// Pipeline: GAT layer (VALU gemm K=10 + edge softmax passes with f32 atomics)
//  -> BN+ReLU -> WMMA f16 GEMM (K=256, LDS-tiled) -> GAT layer (H=1) -> BN+ReLU
//  -> concat climber embedding -> WMMA f16 GEMM (K=128) + bias + ReLU
//  -> tiny FC (K=64, 4 outputs).
// ---------------------------------------------------------------------------

typedef __attribute__((ext_vector_type(16))) _Float16 v16h;
typedef __attribute__((ext_vector_type(8)))  _Float16 v8h;
typedef __attribute__((ext_vector_type(4)))  _Float16 v4h;
typedef __attribute__((ext_vector_type(8)))  float    v8f;

#define BN_EPS 1e-5f

// ---- monotone float<->uint encoding so atomicMax(u32) == float max ---------
__device__ __forceinline__ unsigned fenc(float x) {
    unsigned u = __float_as_uint(x);
    return (u & 0x80000000u) ? ~u : (u | 0x80000000u);
}
__device__ __forceinline__ float fdec(unsigned k) {
    return (k & 0x80000000u) ? __uint_as_float(k & 0x7FFFFFFFu)
                             : __uint_as_float(~k);
}
// fenc(-inf) == 0x007FFFFF (used as init value for segment-max buffers)

// ---------------------------------------------------------------------------
__global__ void fill_u32_k(unsigned* __restrict__ p, long long n, unsigned v) {
    long long i = (long long)blockIdx.x * blockDim.x + threadIdx.x;
    if (i < n) p[i] = v;
}

// h1 = x @ W1   (K = 10, C = 256)
__global__ void gemm_xw1_k(const float* __restrict__ x, const float* __restrict__ W1,
                           float* __restrict__ H1, int n) {
    int idx = blockIdx.x * blockDim.x + threadIdx.x;
    if (idx >= n * 256) return;
    int r = idx >> 8, c = idx & 255;
    float s = 0.f;
    #pragma unroll
    for (int k = 0; k < 10; ++k) s += x[r * 10 + k] * W1[k * 256 + c];
    H1[idx] = s;
}

// e_src[n,h] = <H[n,h,:], a_src[h,:]>,  e_dst likewise
__global__ void attn_scores_k(const float* __restrict__ H, const float* __restrict__ a_src,
                              const float* __restrict__ a_dst, float* __restrict__ es,
                              float* __restrict__ ed, int n, int heads, int hid) {
    int idx = blockIdx.x * blockDim.x + threadIdx.x;
    if (idx >= n * heads) return;
    int node = idx / heads, h = idx % heads;
    const float* hp = H + (size_t)node * heads * hid + (size_t)h * hid;
    float s = 0.f, d = 0.f;
    for (int k = 0; k < hid; ++k) {
        float v = hp[k];
        s += v * a_src[h * hid + k];
        d += v * a_dst[h * hid + k];
    }
    es[idx] = s;
    ed[idx] = d;
}

// pass A: logits (leaky relu 0.2), store to el, atomicMax into segment max
__global__ void edge_pass_a_k(const int* __restrict__ ei, int E_, int n,
                              const float* __restrict__ es, const float* __restrict__ ed,
                              float* __restrict__ el, unsigned* __restrict__ smax, int heads) {
    int idx = blockIdx.x * blockDim.x + threadIdx.x;
    int total = (E_ + n) * heads;
    if (idx >= total) return;
    int e = idx / heads, h = idx % heads;
    int s, d;
    if (e < E_) { s = ei[e]; d = ei[E_ + e]; } else { s = d = e - E_; }
    float v = es[s * heads + h] + ed[d * heads + h];
    v = v > 0.f ? v : 0.2f * v;
    el[idx] = v;
    atomicMax(&smax[d * heads + h], fenc(v));
}

// pass B: w = exp(el - max), overwrite el with w, atomicAdd into segment sum
__global__ void edge_pass_b_k(const int* __restrict__ ei, int E_, int n,
                              float* __restrict__ el, const unsigned* __restrict__ smax,
                              float* __restrict__ ssum, int heads) {
    int idx = blockIdx.x * blockDim.x + threadIdx.x;
    int total = (E_ + n) * heads;
    if (idx >= total) return;
    int e = idx / heads, h = idx % heads;
    int d = (e < E_) ? ei[E_ + e] : (e - E_);
    float w = expf(el[idx] - fdec(smax[d * heads + h]));
    el[idx] = w;
    atomicAdd(&ssum[d * heads + h], w);
}

// pass C: OUT[dst] += alpha * H[src]; hid threads per edge, loop over heads
__global__ void edge_pass_c_k(const int* __restrict__ ei, int E_, int n,
                              const float* __restrict__ el, const float* __restrict__ ssum,
                              const float* __restrict__ H, float* __restrict__ OUT,
                              int heads, int hid) {
    long long idx = (long long)blockIdx.x * blockDim.x + threadIdx.x;
    long long total = (long long)(E_ + n) * hid;
    if (idx >= total) return;
    int e = (int)(idx / hid), t = (int)(idx % hid);
    int s, d;
    if (e < E_) { s = ei[e]; d = ei[E_ + e]; } else { s = d = e - E_; }
    int C = heads * hid;
    for (int h = 0; h < heads; ++h) {
        float alpha = el[(size_t)e * heads + h] / (ssum[d * heads + h] + 1e-16f);
        int col = h * hid + t;
        atomicAdd(&OUT[(size_t)d * C + col], H[(size_t)s * C + col] * alpha);
    }
}

// per-column mean / biased variance (one block per column)
__global__ void bn_stats_k(const float* __restrict__ X, int n, int C,
                           float* __restrict__ mu, float* __restrict__ var) {
    __shared__ float ssum[256], ssq[256];
    int c = blockIdx.x, tid = threadIdx.x;
    float s = 0.f, q = 0.f;
    for (int i = tid; i < n; i += blockDim.x) {
        float v = X[(size_t)i * C + c];
        s += v; q += v * v;
    }
    ssum[tid] = s; ssq[tid] = q;
    __syncthreads();
    for (int st = 128; st > 0; st >>= 1) {
        if (tid < st) { ssum[tid] += ssum[tid + st]; ssq[tid] += ssq[tid + st]; }
        __syncthreads();
    }
    if (tid == 0) {
        float m = ssum[0] / (float)n;
        mu[c] = m;
        var[c] = ssq[0] / (float)n - m * m;
    }
}

// D[r*dstStride + c] = relu((X[r,c]-mu)*rsqrt(var+eps)*g + be)
// (GAT bias cancels exactly under BN: (x+b) - mean(x+b) = x - mean(x))
__global__ void bn_norm_relu_k(const float* __restrict__ X, const float* __restrict__ mu,
                               const float* __restrict__ var, const float* __restrict__ g,
                               const float* __restrict__ be, float* __restrict__ D,
                               int n, int C, int dstStride) {
    int idx = blockIdx.x * blockDim.x + threadIdx.x;
    if (idx >= n * C) return;
    int r = idx / C, c = idx % C;
    float v = (X[idx] - mu[c]) * rsqrtf(var[c] + BN_EPS) * g[c] + be[c];
    D[(size_t)r * dstStride + c] = fmaxf(v, 0.f);
}

// ---------------------------------------------------------------------------
// LDS-tiled WMMA GEMM: D[M,64] = A[M,K] @ B[K,64] (+bias, relu), f16/f32 acc.
// Block = 256 threads = 8 waves -> 32x64 output tile (2x4 wave-tiles of 16x16).
// A tile (32xK) and B (Kx64, transposed) staged in LDS as f16 with +8 halfword
// row padding (lane-to-lane stride = 4 dwords mod 64 banks -> conflict-free
// ds_load_b128 fragment reads).
// Fragment layouts per CDNA5 ISA §7.12.2 (wave32):
//   A 16x32 f16: lane l -> row l%16; K chunk = kb + (l/16)*8 + {0..7}, +16.
//   B 32x16 f16: lane l -> col l%16; K chunk = kb + (l/16)*8 + {0..7}, +16.
//   C/D 16x16 f32: vgpr v -> row v + 8*(l/16), col l%16.
// K is a template param: K/32 WMMAs fully unrolled; EXEC all-ones at every
// WMMA (only post-WMMA stores are predicated).
// ---------------------------------------------------------------------------
template <int K, bool BIAS, bool RELU>
__global__ void __launch_bounds__(256)
wmma_gemm_k(const float* __restrict__ A, const float* __restrict__ B,
            float* __restrict__ D, int M, const float* __restrict__ bias) {
    constexpr int KP = K + 8;                 // padded LDS row stride (f16)
    __shared__ _Float16 As[32 * KP];          // A tile: 32 rows x K
    __shared__ _Float16 Bs[64 * KP];          // B transposed: 64 cols x K

    const int rowBase = blockIdx.x * 32;
    const int tid = threadIdx.x;

    // ---- stage A (coalesced float4 loads, f16 convert, b64 LDS stores) ----
    for (int i = tid; i < 32 * (K / 4); i += 256) {
        int idx4 = i * 4;
        int row = idx4 / K, k = idx4 % K;
        int grow = rowBase + row;
        float4 v = make_float4(0.f, 0.f, 0.f, 0.f);
        if (grow < M) v = *(const float4*)(A + (size_t)grow * K + k);
        v4h h = {(_Float16)v.x, (_Float16)v.y, (_Float16)v.z, (_Float16)v.w};
        *(v4h*)(&As[row * KP + k]) = h;
    }
    // ---- stage B transposed (coalesced float4 loads, b16 LDS stores) ----
    for (int i = tid; i < (K * 64) / 4; i += 256) {
        int idx4 = i * 4;
        int k = idx4 >> 6, c = idx4 & 63;
        float4 v = *(const float4*)(B + (size_t)k * 64 + c);
        Bs[(c + 0) * KP + k] = (_Float16)v.x;
        Bs[(c + 1) * KP + k] = (_Float16)v.y;
        Bs[(c + 2) * KP + k] = (_Float16)v.z;
        Bs[(c + 3) * KP + k] = (_Float16)v.w;
    }
    __syncthreads();

    // ---- per-wave 16x16 tile ----
    const int w = tid >> 5, lane = tid & 31;
    const int tmL = w >> 2, tn = w & 3;
    const int lHalf = lane >> 4, lRow = lane & 15;
    const _Float16* Ap = &As[(tmL * 16 + lRow) * KP];
    const _Float16* Bp = &Bs[(tn * 16 + lRow) * KP];

    v8f acc = {};
    #pragma unroll
    for (int kb = 0; kb < K; kb += 32) {
        v8h a0 = *(const v8h*)(Ap + kb + lHalf * 8);
        v8h a1 = *(const v8h*)(Ap + kb + 16 + lHalf * 8);
        v8h b0 = *(const v8h*)(Bp + kb + lHalf * 8);
        v8h b1 = *(const v8h*)(Bp + kb + 16 + lHalf * 8);
        v16h av = __builtin_shufflevector(a0, a1, 0, 1, 2, 3, 4, 5, 6, 7,
                                          8, 9, 10, 11, 12, 13, 14, 15);
        v16h bv = __builtin_shufflevector(b0, b1, 0, 1, 2, 3, 4, 5, 6, 7,
                                          8, 9, 10, 11, 12, 13, 14, 15);
        acc = __builtin_amdgcn_wmma_f32_16x16x32_f16(false, av, false, bv,
                                                     (short)0, acc, false, false);
    }

    // ---- epilogue (straight-line; guards only affect stores) ----
    const int bcol = tn * 16 + lRow;
    float bsv = 0.f;
    if (BIAS) bsv = bias[bcol];
    const int tileRow = rowBase + tmL * 16;
    #pragma unroll
    for (int v = 0; v < 8; ++v) {
        int row = tileRow + lHalf * 8 + v;
        float val = acc[v] + bsv;
        if (RELU) val = fmaxf(val, 0.f);
        if (row < M) D[(size_t)row * 64 + bcol] = val;
    }
}

// c = relu(climber @ Wc + bc)   (K = 6, C = 64)
__global__ void climber_mlp_k(const float* __restrict__ cl, const float* __restrict__ Wc,
                              const float* __restrict__ bc, float* __restrict__ out, int g) {
    int idx = blockIdx.x * blockDim.x + threadIdx.x;
    if (idx >= g * 64) return;
    int r = idx >> 6, j = idx & 63;
    float s = bc[j];
    #pragma unroll
    for (int k = 0; k < 6; ++k) s += cl[r * 6 + k] * Wc[k * 64 + j];
    out[idx] = fmaxf(s, 0.f);
}

// Z[n, 64+j] = c[batch[n], j]   (Z row stride = 128)
__global__ void gather_c_k(const float* __restrict__ cb, const int* __restrict__ batch,
                           float* __restrict__ Z, int n) {
    int idx = blockIdx.x * blockDim.x + threadIdx.x;
    if (idx >= n * 64) return;
    int r = idx >> 6, j = idx & 63;
    Z[(size_t)r * 128 + 64 + j] = cb[(size_t)batch[r] * 64 + j];
}

// out = ZH @ Wcl2 + bcl2   (K = 64, 4 outputs)
__global__ void final_fc_k(const float* __restrict__ ZH, const float* __restrict__ W,
                           const float* __restrict__ b, float* __restrict__ out, int n) {
    int idx = blockIdx.x * blockDim.x + threadIdx.x;
    if (idx >= n * 4) return;
    int r = idx >> 2, o = idx & 3;
    float s = b[o];
    #pragma unroll
    for (int k = 0; k < 64; ++k) s += ZH[(size_t)r * 64 + k] * W[k * 4 + o];
    out[idx] = s;
}

// ---------------------------------------------------------------------------
static inline int nblk(long long n, int b) { return (int)((n + b - 1) / b); }

extern "C" void kernel_launch(void* const* d_in, const int* in_sizes, int n_in,
                              void* d_out, int out_size, void* d_ws, size_t ws_size,
                              hipStream_t stream) {
    (void)n_in; (void)out_size; (void)ws_size;

    const float* x       = (const float*)d_in[0];
    const float* climber = (const float*)d_in[1];
    const int*   ei      = (const int*)  d_in[2];
    const int*   batch   = (const int*)  d_in[3];
    const float* W1      = (const float*)d_in[4];
    const float* a1_src  = (const float*)d_in[5];
    const float* a1_dst  = (const float*)d_in[6];
    // d_in[7] = b1, d_in[13] = b2  (GAT biases cancel exactly under BN)
    const float* g1      = (const float*)d_in[8];
    const float* be1     = (const float*)d_in[9];
    const float* W2      = (const float*)d_in[10];
    const float* a2_src  = (const float*)d_in[11];
    const float* a2_dst  = (const float*)d_in[12];
    const float* g2      = (const float*)d_in[14];
    const float* be2     = (const float*)d_in[15];
    const float* Wc      = (const float*)d_in[16];
    const float* bc      = (const float*)d_in[17];
    const float* Wcl1    = (const float*)d_in[18];
    const float* bcl1    = (const float*)d_in[19];
    const float* Wcl2    = (const float*)d_in[20];
    const float* bcl2    = (const float*)d_in[21];
    float* out = (float*)d_out;

    const int N  = in_sizes[3];          // nodes
    const int E  = in_sizes[2] / 2;      // edges (before self loops)
    const int G  = in_sizes[1] / 6;      // graphs
    const int EE = E + N;                // edges incl. self loops

    // ---- workspace carve-out ----
    float* ws = (float*)d_ws;
    size_t o = 0;
    float*    H1   = ws + o; o += (size_t)N * 256;   // h1 / normalized h1 / Z
    float*    OUT1 = ws + o; o += (size_t)N * 256;   // aggregate / H2 / OUT2 / ZH
    float*    EL   = ws + o; o += (size_t)EE * 4;    // edge logits -> exp weights
    float*    ES   = ws + o; o += (size_t)N * 4;     // e_src
    float*    ED   = ws + o; o += (size_t)N * 4;     // e_dst
    unsigned* SMAX = (unsigned*)(ws + o); o += (size_t)N * 4;
    float*    SSUM = ws + o; o += (size_t)N * 4;
    float*    CB   = ws + o; o += (size_t)G * 64;    // climber embeddings
    float*    MU   = ws + o; o += 256;
    float*    VAR  = ws + o; o += 256;

    float* H2   = OUT1;                  // [N,64]   after layer-2 GEMM
    float* OUT2 = OUT1 + (size_t)N * 64; // [N,64]   layer-2 aggregate
    float* Z    = H1;                    // [N,128]  concat input
    float* ZH   = OUT1;                  // [N,64]   after Wcl1

    const int B = 256;
    const unsigned ENC_NEG_INF = 0x007FFFFFu;  // fenc(-inf)

    // ========== GAT layer 1 (heads=4, hid=64) ==========
    fill_u32_k<<<nblk((long long)N * 256, B), B, 0, stream>>>((unsigned*)OUT1, (long long)N * 256, 0u);
    gemm_xw1_k<<<nblk((long long)N * 256, B), B, 0, stream>>>(x, W1, H1, N);
    attn_scores_k<<<nblk((long long)N * 4, B), B, 0, stream>>>(H1, a1_src, a1_dst, ES, ED, N, 4, 64);
    fill_u32_k<<<nblk((long long)N * 4, B), B, 0, stream>>>(SMAX, (long long)N * 4, ENC_NEG_INF);
    fill_u32_k<<<nblk((long long)N * 4, B), B, 0, stream>>>((unsigned*)SSUM, (long long)N * 4, 0u);
    edge_pass_a_k<<<nblk((long long)EE * 4, B), B, 0, stream>>>(ei, E, N, ES, ED, EL, SMAX, 4);
    edge_pass_b_k<<<nblk((long long)EE * 4, B), B, 0, stream>>>(ei, E, N, EL, SMAX, SSUM, 4);
    edge_pass_c_k<<<nblk((long long)EE * 64, B), B, 0, stream>>>(ei, E, N, EL, SSUM, H1, OUT1, 4, 64);
    bn_stats_k<<<256, 256, 0, stream>>>(OUT1, N, 256, MU, VAR);
    bn_norm_relu_k<<<nblk((long long)N * 256, B), B, 0, stream>>>(OUT1, MU, VAR, g1, be1, H1, N, 256, 256);

    // ========== layer-2 GEMM: H2 = H1 @ W2 (WMMA f16, K=256, LDS-tiled) ====
    wmma_gemm_k<256, false, false><<<nblk(N, 32), 256, 0, stream>>>(H1, W2, H2, N, nullptr);

    // ========== GAT layer 2 (heads=1, hid=64) ==========
    attn_scores_k<<<nblk((long long)N, B), B, 0, stream>>>(H2, a2_src, a2_dst, ES, ED, N, 1, 64);
    fill_u32_k<<<nblk((long long)N, B), B, 0, stream>>>(SMAX, (long long)N, ENC_NEG_INF);
    fill_u32_k<<<nblk((long long)N, B), B, 0, stream>>>((unsigned*)SSUM, (long long)N, 0u);
    fill_u32_k<<<nblk((long long)N * 64, B), B, 0, stream>>>((unsigned*)OUT2, (long long)N * 64, 0u);
    edge_pass_a_k<<<nblk((long long)EE, B), B, 0, stream>>>(ei, E, N, ES, ED, EL, SMAX, 1);
    edge_pass_b_k<<<nblk((long long)EE, B), B, 0, stream>>>(ei, E, N, EL, SMAX, SSUM, 1);
    edge_pass_c_k<<<nblk((long long)EE * 64, B), B, 0, stream>>>(ei, E, N, EL, SSUM, H2, OUT2, 1, 64);
    bn_stats_k<<<64, 256, 0, stream>>>(OUT2, N, 64, MU, VAR);
    bn_norm_relu_k<<<nblk((long long)N * 64, B), B, 0, stream>>>(OUT2, MU, VAR, g2, be2, Z, N, 64, 128);

    // ========== climber branch + concat ==========
    climber_mlp_k<<<nblk((long long)G * 64, B), B, 0, stream>>>(climber, Wc, bc, CB, G);
    gather_c_k<<<nblk((long long)N * 64, B), B, 0, stream>>>(CB, batch, Z, N);

    // ========== classifier: ZH = relu(Z @ Wcl1 + bcl1) (WMMA f16, K=128) ====
    wmma_gemm_k<128, true, true><<<nblk(N, 32), 256, 0, stream>>>(Z, Wcl1, ZH, N, bcl1);

    // ========== final FC: out = ZH @ Wcl2 + bcl2 ==========
    final_fc_k<<<nblk((long long)N * 4, B), B, 0, stream>>>(ZH, Wcl2, bcl2, out, N);
}
